// Shift_gcn_91242285236821
// MI455X (gfx1250) — matrix operand order repaired and checked by reading, not compiled, not run on hardware
//
#include <hip/hip_runtime.h>
#include <hip/hip_bf16.h>
#include <math.h>

// ---------------- problem constants ----------------
#define PN 32
#define PC 128
#define PT 300
#define PV 19
#define PCO 128
#define PD1 256

typedef __bf16 bf16;
typedef __attribute__((ext_vector_type(16))) __bf16 v16bf;
typedef __attribute__((ext_vector_type(8)))  float  v8f;

#define WMMA_BF16(a, b, c) \
  __builtin_amdgcn_wmma_f32_16x16x32_bf16(false, (a), false, (b), (short)0, (c), false, false)

// LDS leading dims (bf16 elements), padded to dodge bank conflicts
#define LDK1 136   // K=128 rows
#define LDK2 264   // K=256 rows
#define LDT  40    // transposed buffers / g (32 cols + pad)

// ---------------- workspace layout (bytes) ----------------
// bf16 weights, N-major [O][K]
constexpr size_t oWLIN  = 0;                       // [128][128]
constexpr size_t oG1W   = oWLIN  + 16384 * 2;      // [256][128]
constexpr size_t oG2W   = oG1W   + 32768 * 2;      // [256][128]
constexpr size_t oS1W   = oG2W   + 32768 * 2;      // [128][128]
constexpr size_t oS1W1  = oS1W   + 16384 * 2;      // [128][128]
constexpr size_t oS2W   = oS1W1  + 16384 * 2;      // [256][128]
constexpr size_t oS2W1  = oS2W   + 32768 * 2;      // [256][128]
constexpr size_t oS3W   = oS2W1  + 32768 * 2;      // [128][256]
constexpr size_t oS3W1  = oS3W   + 32768 * 2;      // [128][256]
// folded BN affine (fp32)
constexpr size_t oBN1SC = oS3W1  + 32768 * 2;      // [2432]
constexpr size_t oBN1SH = oBN1SC + 2432 * 4;
constexpr size_t oS1SC  = oBN1SH + 2432 * 4;       // [128]
constexpr size_t oS1SH  = oS1SC  + 128 * 4;
constexpr size_t oS2SC  = oS1SH  + 128 * 4;        // [256]
constexpr size_t oS2SH  = oS2SC  + 256 * 4;
constexpr size_t oS3SC  = oS2SH  + 256 * 4;        // [128]
constexpr size_t oS3SH  = oS3SC  + 128 * 4;
// precomputed mask (tanh folded) + fused gather offsets
constexpr size_t oMASK  = oS3SH  + 128 * 4;        // float [2432]: tanh(fm)+1
constexpr size_t oIDXIN = oMASK  + 2432 * 4;       // int [2432]: cs*T*V + vs
constexpr size_t oIDXRS = oIDXIN + 2432 * 4;       // int [2432]: c*T*V + v

// ---------------- fragment loaders (per CDNA5 ISA 16-bit layouts) ------------
// A 16x32: lane L -> row m = mBase + (L&15); hi = L>>4;
//   elems e<8 -> k = kBase + 8*hi + e ;  e>=8 -> k = kBase + 8*hi + 8 + e
__device__ __forceinline__ v16bf load_a_frag(const bf16* base, int ld, int mBase,
                                             int kBase, int lane) {
  const int m  = mBase + (lane & 15);
  const int hi = lane >> 4;
  const bf16* p = base + (size_t)m * ld + kBase + 8 * hi;
  v16bf f;
#pragma unroll
  for (int e = 0; e < 8; ++e)  f[e] = p[e];
#pragma unroll
  for (int e = 8; e < 16; ++e) f[e] = p[e + 8];
  return f;
}

// B 32x16 from N-major storage (row n holds K contiguous):
//   lane L -> col n = nBase + (L&15); hi = L>>4;
//   elems e<8 -> k = kBase + 8*hi + e ;  e>=8 -> k = kBase + 8*hi + 8 + e
__device__ __forceinline__ v16bf load_b_frag(const bf16* base, int ld, int nBase,
                                             int kBase, int lane) {
  const int n  = nBase + (lane & 15);
  const int hi = lane >> 4;
  const bf16* p = base + (size_t)n * ld + kBase + 8 * hi;
  v16bf f;
#pragma unroll
  for (int e = 0; e < 8; ++e)  f[e] = p[e];
#pragma unroll
  for (int e = 8; e < 16; ++e) f[e] = p[e + 8];
  return f;
}

// ---------------- weight prep: fp32 -> bf16 (N-major) + BN folding ----------
__device__ __forceinline__ void bn_affine(const float* p, int cn, int i,
                                          float* sc, float* sh) {
  const float g = p[i], b = p[cn + i], m = p[2 * cn + i], v = p[3 * cn + i];
  const float s = g * rsqrtf(v + 1e-5f);
  sc[i] = s;
  sh[i] = b - m * s;
}

__global__ void prep_kernel(const float* __restrict__ lw,
                            const float* __restrict__ g1w, const float* __restrict__ g2w,
                            const float* __restrict__ s1w, const float* __restrict__ s1w1,
                            const float* __restrict__ s2w, const float* __restrict__ s2w1,
                            const float* __restrict__ s3w, const float* __restrict__ s3w1,
                            const float* __restrict__ bn1p, const float* __restrict__ s1bn,
                            const float* __restrict__ s2bn, const float* __restrict__ s3bn,
                            const float* __restrict__ fmask,
                            const int* __restrict__ shift_in,
                            char* __restrict__ ws) {
  bf16* WlinT = (bf16*)(ws + oWLIN);
  bf16* G1  = (bf16*)(ws + oG1W);
  bf16* G2  = (bf16*)(ws + oG2W);
  bf16* S1  = (bf16*)(ws + oS1W);
  bf16* S11 = (bf16*)(ws + oS1W1);
  bf16* S2  = (bf16*)(ws + oS2W);
  bf16* S21 = (bf16*)(ws + oS2W1);
  bf16* S3  = (bf16*)(ws + oS3W);
  bf16* S31 = (bf16*)(ws + oS3W1);
  float* b1sc = (float*)(ws + oBN1SC);
  float* b1sh = (float*)(ws + oBN1SH);
  float* a1sc = (float*)(ws + oS1SC);
  float* a1sh = (float*)(ws + oS1SH);
  float* a2sc = (float*)(ws + oS2SC);
  float* a2sh = (float*)(ws + oS2SH);
  float* a3sc = (float*)(ws + oS3SC);
  float* a3sh = (float*)(ws + oS3SH);
  float* maskv = (float*)(ws + oMASK);
  int*   idxin = (int*)(ws + oIDXIN);
  int*   idxrs = (int*)(ws + oIDXRS);

  const int i = blockIdx.x * 256 + threadIdx.x;   // 32768 threads
  if (i < 16384) {
    const int o = i >> 7, c = i & 127;
    WlinT[i] = (bf16)lw[c * PCO + o];   // transpose [C][CO] -> [CO][C]
    S1[i]  = (bf16)s1w[i];
    S11[i] = (bf16)s1w1[i];
  }
  if (i < 32768) {
    G1[i]  = (bf16)g1w[i];
    G2[i]  = (bf16)g2w[i];
    S2[i]  = (bf16)s2w[i];
    S21[i] = (bf16)s2w1[i];
    S3[i]  = (bf16)s3w[i];
    S31[i] = (bf16)s3w1[i];
  }
  if (i < PV * PCO) {
    bn_affine(bn1p, PV * PCO, i, b1sc, b1sh);
    // mask with tanh folded once (loop-invariant across all 9600 frames)
    maskv[i] = tanhf(fmask[i]) + 1.0f;
    // fused x0 gather offset for shift_in:  cs*T*V + vs
    const int src = shift_in[i];
    idxin[i] = (src % PC) * (PT * PV) + (src / PC);
    // fused x0 residual offset: c*T*V + v   (j = v*CO + c)
    idxrs[i] = (i % PCO) * (PT * PV) + (i / PCO);
  }
  if (i < PCO) { bn_affine(s1bn, PCO, i, a1sc, a1sh); bn_affine(s3bn, PCO, i, a3sc, a3sh); }
  if (i < PD1)  bn_affine(s2bn, PD1, i, a2sc, a2sh);
}

// ---------------- fused per-frame kernel -------------------------------------
__global__ __launch_bounds__(256)
void shiftgcn_frame_kernel(const float* __restrict__ x0,
                           const float* __restrict__ lbias,
                           const float* __restrict__ g1b, const float* __restrict__ g2b,
                           const float* __restrict__ s1b1, const float* __restrict__ s2b1,
                           const float* __restrict__ s3b1,
                           const int* __restrict__ shift_out,
                           const char* __restrict__ ws,
                           float* __restrict__ out) {
  const bf16* WlinT = (const bf16*)(ws + oWLIN);
  const bf16* G1w   = (const bf16*)(ws + oG1W);
  const bf16* G2w   = (const bf16*)(ws + oG2W);
  const bf16* Wa[3] = {(const bf16*)(ws + oS1W),  (const bf16*)(ws + oS2W),  (const bf16*)(ws + oS3W)};
  const bf16* Wb[3] = {(const bf16*)(ws + oS1W1), (const bf16*)(ws + oS2W1), (const bf16*)(ws + oS3W1)};
  const float* bn1sc = (const float*)(ws + oBN1SC);
  const float* bn1sh = (const float*)(ws + oBN1SH);
  const float* Sc[3] = {(const float*)(ws + oS1SC), (const float*)(ws + oS2SC), (const float*)(ws + oS3SC)};
  const float* Sh[3] = {(const float*)(ws + oS1SH), (const float*)(ws + oS2SH), (const float*)(ws + oS3SH)};
  const float* Bb[3] = {s1b1, s2b1, s3b1};
  const float* maskv = (const float*)(ws + oMASK);
  const int*   idxin = (const int*)(ws + oIDXIN);
  const int*   idxrs = (const int*)(ws + oIDXRS);
  const int Kin[3]   = {128, 128, 256};
  const int NoutS[3] = {128, 256, 128};
  const int xldin[3] = {LDK1, LDK1, LDK2};
  const int xldout[3]= {LDK1, LDK2, 0};

  __shared__ bf16 sIn[32 * LDK1];    // gathered/masked input  [V][C]
  __shared__ bf16 sX [32 * LDK2];    // current activation     [V][K]
  __shared__ bf16 sXT[256 * LDT];    // activation transposed  [K][Vpad]
  __shared__ bf16 sA [32 * LDK2];    // attention 'a'          [V][D1]
  __shared__ bf16 sB [32 * LDK2];    // attention 'b^T'        [V][D1]
  __shared__ bf16 sY [32 * LDK2];    // y = g @ x              [V][K]
  __shared__ bf16 sG [32 * LDT];     // softmax(g) bf16        [V][Vpad]
  __shared__ float sF[32 * LDK1];    // fp32 scratch (lin out / raw g)

  const int tid  = threadIdx.x;
  const int lane = tid & 31;
  const int w    = tid >> 5;
  const int hi   = lane >> 4;
  const int nfrm = blockIdx.x;
  const int nn_b = nfrm / PT;        // batch index
  const int tt_b = nfrm % PT;        // frame index
  // base offset of this frame inside x0 / out: n*C*T*V + t*V
  const size_t baseNT = (size_t)nn_b * (PC * PT * PV) + (size_t)tt_b * PV;

  // warm later-stage weights into L2 (global_prefetch_b8 path)
  {
    const size_t off = (size_t)tid * 256;
    if (off < 65536) {
      __builtin_prefetch((const char*)Wa[1] + off, 0, 1);
      __builtin_prefetch((const char*)Wb[1] + off, 0, 1);
      __builtin_prefetch((const char*)Wa[2] + off, 0, 1);
      __builtin_prefetch((const char*)Wb[2] + off, 0, 1);
    }
  }

  // ---- P0: zero pads, fused gather (shift_in) + mask -> sIn (bf16) ----
  for (int i = tid; i < 32 * LDK1; i += 256) sIn[i] = (bf16)0.0f;
  for (int i = tid; i < 32 * LDK2; i += 256) sX[i]  = (bf16)0.0f;
  for (int i = tid; i < 256 * LDT; i += 256) sXT[i] = (bf16)0.0f;
  __syncthreads();
  for (int j = tid; j < PV * PC; j += 256) {
    const int v = j >> 7, c = j & 127;
    const float xv = x0[baseNT + idxin[j]];
    sIn[v * LDK1 + c] = (bf16)(xv * maskv[j]);
  }
  __syncthreads();

  // ---- P1: lin = xin @ WlinT^T  -> sF fp32 [V][128] (2x8 tiles, 2/wave) ----
  for (int i = 0; i < 2; ++i) {
    const int tt = w * 2 + i;
    const int mT = tt & 1, nT = tt >> 1;
    v8f acc = {};
#pragma unroll
    for (int kt = 0; kt < 4; ++kt) {
      v16bf a = load_a_frag(sIn, LDK1, mT * 16, kt * 32, lane);
      v16bf b = load_b_frag(WlinT, 128, nT * 16, kt * 32, lane);
      acc = WMMA_BF16(a, b, acc);
    }
    const int nn = nT * 16 + (lane & 15);
    const float bias = lbias[nn];
#pragma unroll
    for (int r = 0; r < 8; ++r) {
      const int m = mT * 16 + r + 8 * hi;
      sF[m * 128 + nn] = acc[r] + bias;
    }
  }
  __syncthreads();

  // ---- P2: inverse shift + BN1 + residual + relu -> sX [V][128], sXT ----
  // shift_out[j] is directly the flat index into sF (ld == CO == 128)
  for (int j = tid; j < PV * PCO; j += 256) {
    const int v = j >> 7, c = j & 127;
    float val = sF[shift_out[j]] * bn1sc[j] + bn1sh[j];
    val += x0[baseNT + idxrs[j]];
    val = fmaxf(val, 0.0f);
    const bf16 q = (bf16)val;
    sX[v * LDK1 + c]  = q;
    sXT[c * LDT + v]  = q;
  }
  __syncthreads();

  // ---- P3: a = x@g1w^T + g1b ; b^T = x@g2w^T + g2b  (64 tiles, 8/wave) ----
  for (int i = 0; i < 8; ++i) {
    const int q = w * 8 + i;
    const bf16* Wg     = (q < 32) ? G1w : G2w;
    const float* gbias = (q < 32) ? g1b : g2b;
    bf16* dst          = (q < 32) ? sA  : sB;
    const int qq = q & 31, mT = qq & 1, nT = qq >> 1;
    v8f acc = {};
#pragma unroll
    for (int kt = 0; kt < 4; ++kt) {
      v16bf a = load_a_frag(sX, LDK1, mT * 16, kt * 32, lane);
      v16bf b = load_b_frag(Wg, 128, nT * 16, kt * 32, lane);
      acc = WMMA_BF16(a, b, acc);
    }
    const int nn = nT * 16 + (lane & 15);
    const float bv = gbias[nn];
#pragma unroll
    for (int r = 0; r < 8; ++r) {
      const int m = mT * 16 + r + 8 * hi;
      dst[m * LDK2 + nn] = (bf16)(acc[r] + bv);
    }
  }
  __syncthreads();

  // ---- P4: gmat = a @ b  (K=256, 2x2 tiles on waves 0..3) -> sF [V][Vpad] --
  if (w < 4) {
    const int mT = w & 1, nT = w >> 1;
    v8f acc = {};
#pragma unroll
    for (int kt = 0; kt < 8; ++kt) {
      v16bf a = load_a_frag(sA, LDK2, mT * 16, kt * 32, lane);
      v16bf b = load_b_frag(sB, LDK2, nT * 16, kt * 32, lane);
      acc = WMMA_BF16(a, b, acc);
    }
    const int nn = nT * 16 + (lane & 15);
#pragma unroll
    for (int r = 0; r < 8; ++r) {
      const int m = mT * 16 + r + 8 * hi;
      sF[m * LDT + nn] = acc[r];
    }
  }
  __syncthreads();

  // ---- P5: softmax over joints (rows of g), zero pads, -> sG bf16 ----
  if (tid < 32) {
    const int u = tid;
    if (u < PV) {
      float mx = -3.0e38f;
      for (int v = 0; v < PV; ++v) mx = fmaxf(mx, sF[u * LDT + v]);
      float sum = 0.0f;
      for (int v = 0; v < PV; ++v) {
        const float e = expf(sF[u * LDT + v] - mx);
        sF[u * LDT + v] = e;
        sum += e;
      }
      const float inv = 1.0f / sum;
      for (int v = 0; v < 32; ++v)
        sG[u * LDT + v] = (v < PV) ? (bf16)(sF[u * LDT + v] * inv) : (bf16)0.0f;
    } else {
      for (int v = 0; v < 32; ++v) sG[u * LDT + v] = (bf16)0.0f;
    }
  }
  __syncthreads();

  // ---- P6: three sgcn stages -------------------------------------------------
  for (int s = 0; s < 3; ++s) {
    const int K = Kin[s], Nout = NoutS[s];
    const int xld = xldin[s];

    // y = g @ x : A = sG [V][32], B = x^T rows (K-contiguous over joints)
    const int nyt = 2 * (K / 16);       // tiles
    const int py  = nyt / 8;
    for (int i = 0; i < py; ++i) {
      const int tt = w * py + i;
      const int mT = tt & 1, nT = tt >> 1;
      v16bf a = load_a_frag(sG, LDT, mT * 16, 0, lane);
      v16bf b = load_b_frag(sXT, LDT, nT * 16, 0, lane);
      v8f acc = {};
      acc = WMMA_BF16(a, b, acc);
      const int nn = nT * 16 + (lane & 15);
#pragma unroll
      for (int r = 0; r < 8; ++r) {
        const int m = mT * 16 + r + 8 * hi;
        sY[m * LDK2 + nn] = (bf16)acc[r];
      }
    }
    __syncthreads();

    // h = y @ w^T + x @ w1^T  (accumulate both into one C chain)
    const int nht = 2 * (Nout / 16);
    const int ph  = nht / 8;            // 2 or 4
    v8f hacc[4];
    for (int i = 0; i < ph; ++i) {
      const int tt = w * ph + i;
      const int mT = tt & 1, nT = tt >> 1;
      v8f acc = {};
      for (int kt = 0; kt < K / 32; ++kt) {
        v16bf a  = load_a_frag(sY, LDK2, mT * 16, kt * 32, lane);
        v16bf b  = load_b_frag(Wa[s], K, nT * 16, kt * 32, lane);
        acc = WMMA_BF16(a, b, acc);
        v16bf a2 = load_a_frag(sX, xld, mT * 16, kt * 32, lane);
        v16bf b2 = load_b_frag(Wb[s], K, nT * 16, kt * 32, lane);
        acc = WMMA_BF16(a2, b2, acc);
      }
      hacc[i] = acc;
    }
    __syncthreads();   // all reads of sX/sXT complete; safe to overwrite

    // epilogue: +bias, BN affine, relu; write next x / x^T (or global out)
    for (int i = 0; i < ph; ++i) {
      const int tt = w * ph + i;
      const int mT = tt & 1, nT = tt >> 1;
      const int nn = nT * 16 + (lane & 15);
      const float bv = Bb[s][nn], sc = Sc[s][nn], sh = Sh[s][nn];
#pragma unroll
      for (int r = 0; r < 8; ++r) {
        const int m = mT * 16 + r + 8 * hi;
        float val = (hacc[i][r] + bv) * sc + sh;
        val = fmaxf(val, 0.0f);
        if (m >= PV) val = 0.0f;
        if (s < 2) {
          const bf16 q = (bf16)val;
          sX[m * xldout[s] + nn] = q;
          sXT[nn * LDT + m]      = q;
        } else if (m < PV) {
          out[baseNT + (size_t)nn * (PT * PV) + m] = val;
        }
      }
    }
    __syncthreads();
  }
}

// ---------------- host side --------------------------------------------------
extern "C" void kernel_launch(void* const* d_in, const int* in_sizes, int n_in,
                              void* d_out, int out_size, void* d_ws, size_t ws_size,
                              hipStream_t stream) {
  const float* x0    = (const float*)d_in[0];
  const float* fmask = (const float*)d_in[1];
  const float* lw    = (const float*)d_in[2];
  const float* lb    = (const float*)d_in[3];
  const float* bn1p  = (const float*)d_in[4];
  const float* g1w   = (const float*)d_in[5];
  const float* g1b   = (const float*)d_in[6];
  const float* g2w   = (const float*)d_in[7];
  const float* g2b   = (const float*)d_in[8];
  const float* s1w   = (const float*)d_in[9];
  const float* s1w1  = (const float*)d_in[10];
  const float* s1b1  = (const float*)d_in[11];
  const float* s1bn  = (const float*)d_in[12];
  const float* s2w   = (const float*)d_in[13];
  const float* s2w1  = (const float*)d_in[14];
  const float* s2b1  = (const float*)d_in[15];
  const float* s2bn  = (const float*)d_in[16];
  const float* s3w   = (const float*)d_in[17];
  const float* s3w1  = (const float*)d_in[18];
  const float* s3b1  = (const float*)d_in[19];
  const float* s3bn  = (const float*)d_in[20];
  const int* shin    = (const int*)d_in[21];
  const int* shout   = (const int*)d_in[22];
  char* ws = (char*)d_ws;
  float* outp = (float*)d_out;

  // weight conversion + BN folding + gather-offset/mask precompute (one-time)
  prep_kernel<<<128, 256, 0, stream>>>(lw, g1w, g2w, s1w, s1w1, s2w, s2w1,
                                       s3w, s3w1, bn1p, s1bn, s2bn, s3bn,
                                       fmask, shin, ws);

  // one workgroup per (n, t) frame, fully fused
  shiftgcn_frame_kernel<<<PN * PT, 256, 0, stream>>>(
      x0, lb, g1b, g2b, s1b1, s2b1, s3b1, shout, ws, outp);
}
